// Model_22445499089054
// MI455X (gfx1250) — matrix operand order repaired and checked by reading, not compiled, not run on hardware
//
#include <hip/hip_runtime.h>

typedef _Float16 h16;
typedef __attribute__((ext_vector_type(16))) _Float16 v16h;
typedef __attribute__((ext_vector_type(8)))  _Float16 v8h;
typedef __attribute__((ext_vector_type(8)))  float    v8f;

static inline int cdiv(long a, long b) { return (int)((a + b - 1) / b); }
static inline int ilog2(int v) { int l = 0; while ((1 << l) < v) l++; return l; }

// ---------------------------------------------------------------------------
// Weight packing: f32 (tap, cin, co) -> f16 WMMA B-fragment layout:
//   dst[((nb*KB + k)*32 + lane)*16 + t] = W[tap=(k*32+K)/CinP][cin=(k*32+K)%CinP][co=nb*16+(lane&15)]
//   with K = (t>>3)*16 + (lane>>4)*8 + (t&7)   (B 32x16 f16 layout: lanes 0-15 K=0..7,16..23, lanes 16-31 K=8..15,24..31)
// ---------------------------------------------------------------------------
__global__ void pack_w(const float* __restrict__ src, h16* __restrict__ dst,
                       int taps, int CinR, int CinP, int CoR, int CoP) {
    int KB = (taps * CinP) >> 5;
    long total = (long)(CoP >> 4) * KB * 512;
    long i = (long)blockIdx.x * blockDim.x + threadIdx.x;
    if (i >= total) return;
    int t    = (int)(i & 15);
    int lane = (int)((i >> 4) & 31);
    long r   = i >> 9;
    int k  = (int)(r % KB);
    int nb = (int)(r / KB);
    int K  = ((t >> 3) << 4) + ((lane >> 4) << 3) + (t & 7);
    int gk = k * 32 + K;
    int tap = gk / CinP;
    int cp  = gk % CinP;
    int N   = nb * 16 + (lane & 15);
    float v = 0.f;
    if (cp < CinR && N < CoR) v = src[((long)tap * CinR + cp) * CoR + N];
    dst[i] = (h16)v;
}

// 2x2x2 max-pool of occupancy mask (dims are powers of two)
__global__ void pool_mask(const float* __restrict__ mi, float* __restrict__ mo, int lw) {
    int i = blockIdx.x * blockDim.x + threadIdx.x;
    int V = 1 << (3 * lw);
    if (i >= V) return;
    int Wm = (1 << lw) - 1;
    int x = i & Wm, y = (i >> lw) & Wm, z = i >> (2 * lw);
    int lwi = lw + 1;
    float mx = 0.f;
    for (int dz = 0; dz < 2; dz++)
        for (int dy = 0; dy < 2; dy++)
            for (int dx = 0; dx < 2; dx++) {
                float v = mi[((((2 * z + dz) << lwi) + (2 * y + dy)) << lwi) + (2 * x + dx)];
                mx = fmaxf(mx, v);
            }
    mo[i] = mx;
}

// x (V,4) f32 * mask -> f16 tensor with 32 channels (4 real + zero pad)
__global__ void stage_input(const float* __restrict__ x, const float* __restrict__ mask,
                            h16* __restrict__ out, int V) {
    int i = blockIdx.x * blockDim.x + threadIdx.x;
    if (i >= V * 32) return;
    int v = i >> 5, c = i & 31;
    float val = (c < 4) ? x[v * 4 + c] * mask[v] : 0.f;
    out[i] = (h16)val;
}

// copy skip tensor (V,Cs) into channels [off, off+Cs) of concat buffer (V,Ctot)
__global__ void copy_skip(const h16* __restrict__ skip, h16* __restrict__ dst,
                          int V, int Cs, int Ctot, int off) {
    int i = blockIdx.x * blockDim.x + threadIdx.x;
    if (i >= V * Cs) return;
    int v = i / Cs, c = i % Cs;
    dst[(long)v * Ctot + off + c] = skip[i];
}

// ---------------------------------------------------------------------------
// Generic masked conv3d (+BN, +mask, optional residual, optional ReLU) as
// implicit GEMM with v_wmma_f32_16x16x32_f16. One wave: 16 voxels x NACC*16 co.
// The 4 waves of a block share the co-tile; per-tap weight slab staged in LDS.
// All spatial dims are powers of two (lw = log2 out dim, lwi = log2 in dim).
// ---------------------------------------------------------------------------
template <int NACC>
__global__ void conv_wmma(const h16* __restrict__ in, const h16* __restrict__ wp,
                          const float* __restrict__ bnS, const float* __restrict__ bnB,
                          const float* __restrict__ bnM, const float* __restrict__ bnV,
                          const float* __restrict__ mask, const h16* __restrict__ res,
                          void* __restrict__ outp,
                          int lw, int lwi, int Cin, int Co,
                          int ks, int stride, int pad,
                          int relu, int f32out, int CoReal) {
    extern __shared__ h16 ldsb[];
    int tid  = threadIdx.x;
    int wid  = blockIdx.x * (blockDim.x >> 5) + (tid >> 5);
    int lane = tid & 31;
    int lmt = 3 * lw - 4;              // log2(#voxel tiles)
    int mt = wid & ((1 << lmt) - 1);
    int ct = wid >> lmt;
    int f0 = mt << 4;
    int m  = lane & 15, hh = lane >> 4;
    int f = f0 + m;
    int Wm = (1 << lw) - 1;
    int x = f & Wm, y = (f >> lw) & Wm, z = f >> (2 * lw);
    int IWu = 1 << lwi;
    int taps = ks * ks * ks;
    int KBc  = Cin >> 5;               // 32-wide K blocks per tap
    int KB   = taps * KBc;
    int slab = KBc << 6;               // 16B chunks per n-block per tap

    v8h zero8;
#pragma unroll
    for (int q = 0; q < 8; q++) zero8[q] = (h16)0.f;
    v8f acc[NACC];
#pragma unroll
    for (int a = 0; a < NACC; a++)
#pragma unroll
        for (int q = 0; q < 8; q++) acc[a][q] = 0.f;

    int dxt = 0, dyt = 0, dzt = 0;     // incremental tap decode
    for (int kt = 0; kt < taps; kt++) {
        // stage this tap's B slab (NACC n-blocks x KBc k-blocks) into LDS
        __syncthreads();
#pragma unroll
        for (int a = 0; a < NACC; a++) {
            const v8h* gs = (const v8h*)wp +
                ((((long)(ct * NACC + a) * KB + (long)kt * KBc) << 5) << 1);
            v8h* ds = (v8h*)ldsb + a * slab;
            for (int idx = tid; idx < slab; idx += 128) ds[idx] = gs[idx];
        }
        __syncthreads();

        int zi = z * stride + dzt - pad;
        int yi = y * stride + dyt - pad;
        int xi = x * stride + dxt - pad;
        bool ok = ((unsigned)zi < (unsigned)IWu) && ((unsigned)yi < (unsigned)IWu) &&
                  ((unsigned)xi < (unsigned)IWu);
        long off = (long)((((zi << lwi) + yi) << lwi) + xi) * Cin + hh * 8;
        const v16h* lv = (const v16h*)ldsb;
        for (int kb = 0; kb < KBc; kb++) {
            v8h lo = zero8, hi = zero8;
            if (ok) {
                lo = *(const v8h*)(in + off + kb * 32);
                hi = *(const v8h*)(in + off + kb * 32 + 16);
            }
            v16h afrag = __builtin_shufflevector(lo, hi, 0, 1, 2, 3, 4, 5, 6, 7,
                                                 8, 9, 10, 11, 12, 13, 14, 15);
#pragma unroll
            for (int a = 0; a < NACC; a++) {
                v16h bfrag = lv[(a * KBc + kb) * 32 + lane];
                acc[a] = __builtin_amdgcn_wmma_f32_16x16x32_f16(
                    false, afrag, false, bfrag, (short)0, acc[a], false, false);
            }
        }
        if (++dxt == ks) { dxt = 0; if (++dyt == ks) { dyt = 0; ++dzt; } }
    }

    // fused BN + mask + residual + relu store (D layout: N=lane&15, M=r+8*hh)
#pragma unroll
    for (int a = 0; a < NACC; a++) {
        int c = (ct * NACC + a) * 16 + (lane & 15);
        float aS = 1.f, bS = 0.f;
        if (bnS) {
            aS = bnS[c] * rsqrtf(bnV[c] + 1e-5f);
            bS = bnB[c] - bnM[c] * aS;
        } else if (bnB && c < CoReal) {
            bS = bnB[c];
        }
#pragma unroll
        for (int r = 0; r < 8; r++) {
            int fo = f0 + r + 8 * hh;
            float val = acc[a][r] * aS + bS;
            val *= mask[fo];
            if (res) val += (float)res[(long)fo * Co + c];
            if (relu) val = fmaxf(val, 0.f);
            if (f32out) {
                if (c < CoReal) ((float*)outp)[(long)fo * CoReal + c] = val;
            } else {
                ((h16*)outp)[(long)fo * Co + c] = (h16)val;
            }
        }
    }
}

// ---------------------------------------------------------------------------
// ks=2 stride=2 transposed conv (+BN+ReLU+mask) into concat buffer.
// out[2i+d] = in[i] * w[1-d] (jax conv_transpose, transpose_kernel=False).
// Partitioned by output parity so the weight tap is wave-uniform; the single
// tap's B slab is staged into LDS once (shared by the block's 4 waves).
// ---------------------------------------------------------------------------
template <int NACC>
__global__ void deconv_wmma(const h16* __restrict__ in, const h16* __restrict__ wp,
                            const float* __restrict__ bnS, const float* __restrict__ bnB,
                            const float* __restrict__ bnM, const float* __restrict__ bnV,
                            const float* __restrict__ mask, h16* __restrict__ out,
                            int lw, int lct, int Cin, int Co, int Ctot) {
    extern __shared__ h16 ldsb[];
    int tid  = threadIdx.x;
    int wid  = blockIdx.x * (blockDim.x >> 5) + (tid >> 5);
    int lane = tid & 31;
    int lwi = lw - 1;
    int lmt = 3 * lwi - 4;
    int mt   = wid & ((1 << lmt) - 1);
    int rest = wid >> lmt;
    int ct  = rest & ((1 << lct) - 1);
    int par = rest >> lct;
    int px = par & 1, py = (par >> 1) & 1, pz = (par >> 2) & 1;
    int kt = ((pz ^ 1) << 2) | ((py ^ 1) << 1) | (px ^ 1);   // spatial flip
    int v0 = mt << 4;
    int m  = lane & 15, hh = lane >> 4;
    int KBc = Cin >> 5;
    int KB  = 8 * KBc;
    int slab = KBc << 6;

    // stage the single tap's B slab into LDS
#pragma unroll
    for (int a = 0; a < NACC; a++) {
        const v8h* gs = (const v8h*)wp +
            ((((long)(ct * NACC + a) * KB + (long)kt * KBc) << 5) << 1);
        v8h* ds = (v8h*)ldsb + a * slab;
        for (int idx = tid; idx < slab; idx += 128) ds[idx] = gs[idx];
    }
    __syncthreads();

    v8f acc[NACC];
#pragma unroll
    for (int a = 0; a < NACC; a++)
#pragma unroll
        for (int q = 0; q < 8; q++) acc[a][q] = 0.f;

    const h16* base = in + (long)(v0 + m) * Cin + hh * 8;
    const v16h* lv  = (const v16h*)ldsb;
    for (int kb = 0; kb < KBc; kb++) {
        v8h lo = *(const v8h*)(base + kb * 32);
        v8h hi = *(const v8h*)(base + kb * 32 + 16);
        v16h afrag = __builtin_shufflevector(lo, hi, 0, 1, 2, 3, 4, 5, 6, 7,
                                             8, 9, 10, 11, 12, 13, 14, 15);
#pragma unroll
        for (int a = 0; a < NACC; a++) {
            v16h bfrag = lv[(a * KBc + kb) * 32 + lane];
            acc[a] = __builtin_amdgcn_wmma_f32_16x16x32_f16(
                false, afrag, false, bfrag, (short)0, acc[a], false, false);
        }
    }
    int IWm = (1 << lwi) - 1;
#pragma unroll
    for (int a = 0; a < NACC; a++) {
        int c  = (ct * NACC + a) * 16 + (lane & 15);
        float aS = bnS[c] * rsqrtf(bnV[c] + 1e-5f);
        float bS = bnB[c] - bnM[c] * aS;
#pragma unroll
        for (int r = 0; r < 8; r++) {
            int v  = v0 + r + 8 * hh;
            int xi = v & IWm, yi = (v >> lwi) & IWm, zi = v >> (2 * lwi);
            int fo = ((((2 * zi + pz) << lw) + (2 * yi + py)) << lw) + (2 * xi + px);
            float val = acc[a][r] * aS + bS;
            val *= mask[fo];
            val = fmaxf(val, 0.f);
            out[(long)fo * Ctot + c] = (h16)val;
        }
    }
}

// ---------------------------------------------------------------------------
// Host-side orchestration
// ---------------------------------------------------------------------------
struct BN4 { const float *s, *b, *m, *v; };

static void conv_run(hipStream_t st, const h16* in, const h16* wp, BN4 bn,
                     const float* mask, const h16* res, void* out,
                     int lw, int lwi, int Cin, int Co, int ks, int stride, int pad,
                     int relu, int f32out, int CoReal) {
    int mtiles = 1 << (3 * lw - 4);
    int nacc = (Co % 64 == 0) ? 4 : (Co % 48 == 0) ? 3 : 2;
    int cot = Co / (16 * nacc);
    int waves = mtiles * cot;
    size_t lds = (size_t)nacc * (Cin >> 5) * 1024;
    if (nacc == 4)
        conv_wmma<4><<<cdiv(waves, 4), 128, lds, st>>>(in, wp, bn.s, bn.b, bn.m, bn.v,
            mask, res, out, lw, lwi, Cin, Co, ks, stride, pad, relu, f32out, CoReal);
    else if (nacc == 3)
        conv_wmma<3><<<cdiv(waves, 4), 128, lds, st>>>(in, wp, bn.s, bn.b, bn.m, bn.v,
            mask, res, out, lw, lwi, Cin, Co, ks, stride, pad, relu, f32out, CoReal);
    else
        conv_wmma<2><<<cdiv(waves, 4), 128, lds, st>>>(in, wp, bn.s, bn.b, bn.m, bn.v,
            mask, res, out, lw, lwi, Cin, Co, ks, stride, pad, relu, f32out, CoReal);
}

static void deconv_run(hipStream_t st, const h16* in, const h16* wp, BN4 bn,
                       const float* mask, h16* out, int lw, int Cin, int Co, int Ctot) {
    int mtiles = 1 << (3 * (lw - 1) - 4);
    int nacc = (Co % 64 == 0) ? 4 : (Co % 48 == 0) ? 3 : 2;
    int cot = Co / (16 * nacc);
    int lct = ilog2(cot);
    int waves = mtiles * cot * 8;
    size_t lds = (size_t)nacc * (Cin >> 5) * 1024;
    if (nacc == 4)
        deconv_wmma<4><<<cdiv(waves, 4), 128, lds, st>>>(in, wp, bn.s, bn.b, bn.m, bn.v,
            mask, out, lw, lct, Cin, Co, Ctot);
    else if (nacc == 3)
        deconv_wmma<3><<<cdiv(waves, 4), 128, lds, st>>>(in, wp, bn.s, bn.b, bn.m, bn.v,
            mask, out, lw, lct, Cin, Co, Ctot);
    else
        deconv_wmma<2><<<cdiv(waves, 4), 128, lds, st>>>(in, wp, bn.s, bn.b, bn.m, bn.v,
            mask, out, lw, lct, Cin, Co, Ctot);
}

extern "C" void kernel_launch(void* const* d_in, const int* in_sizes, int n_in,
                              void* d_out, int out_size, void* d_ws, size_t ws_size,
                              hipStream_t stream) {
    (void)in_sizes; (void)n_in; (void)out_size; (void)ws_size;
    auto F = [&](int i) -> const float* { return (const float*)d_in[i]; };
    auto bnq = [&](int i) -> BN4 { return BN4{F(i + 2), F(i + 0), F(i + 1), F(i + 3)}; };

    char* wsp = (char*)d_ws;
    auto alloc = [&](size_t b) -> void* {
        void* p = (void*)wsp;
        wsp += (b + 255) & ~(size_t)255;
        return p;
    };

    const float* X  = F(0);
    const float* M0 = F(1);

    const int V0 = 262144, V1 = 32768, V2 = 4096, V3 = 512;

    // pooled masks
    float* M1 = (float*)alloc((size_t)V1 * 4);
    float* M2 = (float*)alloc((size_t)V2 * 4);
    float* M3 = (float*)alloc((size_t)V3 * 4);
    float* M4 = (float*)alloc((size_t)64 * 4);
    pool_mask<<<cdiv(V1, 256), 256, 0, stream>>>(M0, M1, 5);
    pool_mask<<<cdiv(V2, 256), 256, 0, stream>>>(M1, M2, 4);
    pool_mask<<<cdiv(V3, 256), 256, 0, stream>>>(M2, M3, 3);
    pool_mask<<<1, 256, 0, stream>>>(M3, M4, 2);

    // packed weights
    auto prep = [&](int idx, int taps, int cinR, int cinP, int coR, int coP) -> const h16* {
        size_t n = (size_t)coP * taps * cinP;
        h16* dst = (h16*)alloc(n * 2);
        pack_w<<<cdiv((long)n, 256), 256, 0, stream>>>(F(idx), dst, taps, cinR, cinP, coR, coP);
        return dst;
    };
    const h16* Wst1  = prep(127, 27,   4,  32,  32,  32);
    const h16* Wst2  = prep(128, 27,  32,  32,  32,  32);
    const h16* Ws1d  = prep(8,    8,  32,  32,  32,  32);
    const h16* Ws1ra = prep(17,  27,  32,  32,  32,  32);
    const h16* Ws1rb = prep(18,  27,  32,  32,  32,  32);
    const h16* Ws1sa = prep(27,  27,  32,  32,  32,  32);
    const h16* Ws1sb = prep(28,  27,  32,  32,  32,  32);
    const h16* Ws2d  = prep(33,   8,  32,  32,  32,  32);
    const h16* Ws2ra = prep(46,  27,  32,  32,  64,  64);
    const h16* Ws2rb = prep(47,  27,  64,  64,  64,  64);
    const h16* Ws2rd = prep(48,   1,  32,  32,  64,  64);
    const h16* Ws2sa = prep(57,  27,  64,  64,  64,  64);
    const h16* Ws2sb = prep(58,  27,  64,  64,  64,  64);
    const h16* Ws3d  = prep(63,   8,  64,  64,  64,  64);
    const h16* Ws3ra = prep(76,  27,  64,  64, 128, 128);
    const h16* Ws3rb = prep(77,  27, 128, 128, 128, 128);
    const h16* Ws3rd = prep(78,   1,  64,  64, 128, 128);
    const h16* Ws3sa = prep(87,  27, 128, 128, 128, 128);
    const h16* Ws3sb = prep(88,  27, 128, 128, 128, 128);
    const h16* Ws4d  = prep(93,   8, 128, 128, 128, 128);
    const h16* Ws4ra = prep(106, 27, 128, 128, 256, 256);
    const h16* Ws4rb = prep(107, 27, 256, 256, 256, 256);
    const h16* Ws4rd = prep(108,  1, 128, 128, 256, 256);
    const h16* Ws4sa = prep(117, 27, 256, 256, 256, 256);
    const h16* Ws4sb = prep(118, 27, 256, 256, 256, 256);
    const h16* Wu1u  = prep(158,  8, 256, 256, 256, 256);
    const h16* Wu1ra = prep(141, 27, 384, 384, 256, 256);
    const h16* Wu1rb = prep(142, 27, 256, 256, 256, 256);
    const h16* Wu1rd = prep(143,  1, 384, 384, 256, 256);
    const h16* Wu1sa = prep(152, 27, 256, 256, 256, 256);
    const h16* Wu1sb = prep(153, 27, 256, 256, 256, 256);
    const h16* Wu2u  = prep(188,  8, 256, 256, 128, 128);
    const h16* Wu2ra = prep(171, 27, 192, 192, 128, 128);
    const h16* Wu2rb = prep(172, 27, 128, 128, 128, 128);
    const h16* Wu2rd = prep(173,  1, 192, 192, 128, 128);
    const h16* Wu2sa = prep(182, 27, 128, 128, 128, 128);
    const h16* Wu2sb = prep(183, 27, 128, 128, 128, 128);
    const h16* Wu3u  = prep(218,  8, 128, 128,  96,  96);
    const h16* Wu3ra = prep(201, 27, 128, 128,  96,  96);
    const h16* Wu3rb = prep(202, 27,  96,  96,  96,  96);
    const h16* Wu3rd = prep(203,  1, 128, 128,  96,  96);
    const h16* Wu3sa = prep(212, 27,  96,  96,  96,  96);
    const h16* Wu3sb = prep(213, 27,  96,  96,  96,  96);
    const h16* Wu4u  = prep(248,  8,  96,  96,  96,  96);
    const h16* Wu4ra = prep(231, 27, 128, 128,  96,  96);
    const h16* Wu4rb = prep(232, 27,  96,  96,  96,  96);
    const h16* Wu4rd = prep(233,  1, 128, 128,  96,  96);
    const h16* Wu4sa = prep(242, 27,  96,  96,  96,  96);
    const h16* Wu4sb = prep(243, 27,  96,  96,  96,  96);
    const h16* Wcls  = prep(3,    1,  96,  96,  20,  32);

    // activation buffers (f16)
    h16* IN0 = (h16*)alloc((size_t)V0 * 32 * 2);
    h16* X0  = (h16*)alloc((size_t)V0 * 32 * 2);
    h16* CC0 = (h16*)alloc((size_t)V0 * 128 * 2);
    h16* T0a = (h16*)alloc((size_t)V0 * 96 * 2);
    h16* T0b = (h16*)alloc((size_t)V0 * 96 * 2);
    h16* T0c = (h16*)alloc((size_t)V0 * 96 * 2);
    h16* CC1 = (h16*)alloc((size_t)V1 * 128 * 2);
    h16* T1a = (h16*)alloc((size_t)V1 * 96 * 2);
    h16* T1b = (h16*)alloc((size_t)V1 * 96 * 2);
    h16* T1c = (h16*)alloc((size_t)V1 * 96 * 2);
    h16* T1d = (h16*)alloc((size_t)V1 * 96 * 2);
    h16* CC2 = (h16*)alloc((size_t)V2 * 192 * 2);
    h16* T2a = (h16*)alloc((size_t)V2 * 128 * 2);
    h16* T2b = (h16*)alloc((size_t)V2 * 128 * 2);
    h16* T2c = (h16*)alloc((size_t)V2 * 128 * 2);
    h16* T2d = (h16*)alloc((size_t)V2 * 128 * 2);
    h16* CC3 = (h16*)alloc((size_t)V3 * 384 * 2);
    h16* T3a = (h16*)alloc((size_t)V3 * 256 * 2);
    h16* T3b = (h16*)alloc((size_t)V3 * 256 * 2);
    h16* T3c = (h16*)alloc((size_t)V3 * 256 * 2);
    h16* T3d = (h16*)alloc((size_t)V3 * 256 * 2);
    h16* T4a = (h16*)alloc((size_t)64 * 256 * 2);
    h16* T4b = (h16*)alloc((size_t)64 * 256 * 2);
    h16* T4c = (h16*)alloc((size_t)64 * 256 * 2);
    h16* T4d = (h16*)alloc((size_t)64 * 256 * 2);

    // ---- forward ----
    stage_input<<<cdiv((long)V0 * 32, 256), 256, 0, stream>>>(X, M0, IN0, V0);
    // stem
    conv_run(stream, IN0, Wst1, bnq(119), M0, nullptr, T0a, 6, 6, 32, 32, 3, 1, 1, 1, 0, 32);
    conv_run(stream, T0a, Wst2, bnq(123), M0, nullptr, X0, 6, 6, 32, 32, 3, 1, 1, 1, 0, 32);
    // stage1 (32->32, identity shortcut)
    conv_run(stream, X0, Ws1d, bnq(4), M1, nullptr, T1a, 5, 6, 32, 32, 2, 2, 0, 1, 0, 32);
    conv_run(stream, T1a, Ws1ra, bnq(9), M1, nullptr, T1b, 5, 5, 32, 32, 3, 1, 1, 1, 0, 32);
    conv_run(stream, T1b, Ws1rb, bnq(13), M1, T1a, T1c, 5, 5, 32, 32, 3, 1, 1, 1, 0, 32);
    conv_run(stream, T1c, Ws1sa, bnq(19), M1, nullptr, T1a, 5, 5, 32, 32, 3, 1, 1, 1, 0, 32);
    conv_run(stream, T1a, Ws1sb, bnq(23), M1, T1c, T1b, 5, 5, 32, 32, 3, 1, 1, 1, 0, 32);
    h16* x1 = T1b;
    // stage2 (32->64)
    conv_run(stream, x1, Ws2d, bnq(29), M2, nullptr, T2a, 4, 5, 32, 32, 2, 2, 0, 1, 0, 32);
    conv_run(stream, T2a, Ws2rd, bnq(42), M2, nullptr, T2b, 4, 4, 32, 64, 1, 1, 0, 0, 0, 64);
    conv_run(stream, T2a, Ws2ra, bnq(34), M2, nullptr, T2c, 4, 4, 32, 64, 3, 1, 1, 1, 0, 64);
    conv_run(stream, T2c, Ws2rb, bnq(38), M2, T2b, T2d, 4, 4, 64, 64, 3, 1, 1, 1, 0, 64);
    conv_run(stream, T2d, Ws2sa, bnq(49), M2, nullptr, T2a, 4, 4, 64, 64, 3, 1, 1, 1, 0, 64);
    conv_run(stream, T2a, Ws2sb, bnq(53), M2, T2d, T2b, 4, 4, 64, 64, 3, 1, 1, 1, 0, 64);
    h16* x2 = T2b;
    // stage3 (64->128)
    conv_run(stream, x2, Ws3d, bnq(59), M3, nullptr, T3a, 3, 4, 64, 64, 2, 2, 0, 1, 0, 64);
    conv_run(stream, T3a, Ws3rd, bnq(72), M3, nullptr, T3b, 3, 3, 64, 128, 1, 1, 0, 0, 0, 128);
    conv_run(stream, T3a, Ws3ra, bnq(64), M3, nullptr, T3c, 3, 3, 64, 128, 3, 1, 1, 1, 0, 128);
    conv_run(stream, T3c, Ws3rb, bnq(68), M3, T3b, T3d, 3, 3, 128, 128, 3, 1, 1, 1, 0, 128);
    conv_run(stream, T3d, Ws3sa, bnq(79), M3, nullptr, T3a, 3, 3, 128, 128, 3, 1, 1, 1, 0, 128);
    conv_run(stream, T3a, Ws3sb, bnq(83), M3, T3d, T3b, 3, 3, 128, 128, 3, 1, 1, 1, 0, 128);
    h16* x3 = T3b;
    // stage4 (128->256)
    conv_run(stream, x3, Ws4d, bnq(89), M4, nullptr, T4a, 2, 3, 128, 128, 2, 2, 0, 1, 0, 128);
    conv_run(stream, T4a, Ws4rd, bnq(102), M4, nullptr, T4b, 2, 2, 128, 256, 1, 1, 0, 0, 0, 256);
    conv_run(stream, T4a, Ws4ra, bnq(94), M4, nullptr, T4c, 2, 2, 128, 256, 3, 1, 1, 1, 0, 256);
    conv_run(stream, T4c, Ws4rb, bnq(98), M4, T4b, T4d, 2, 2, 256, 256, 3, 1, 1, 1, 0, 256);
    conv_run(stream, T4d, Ws4sa, bnq(109), M4, nullptr, T4a, 2, 2, 256, 256, 3, 1, 1, 1, 0, 256);
    conv_run(stream, T4a, Ws4sb, bnq(113), M4, T4d, T4b, 2, 2, 256, 256, 3, 1, 1, 1, 0, 256);
    h16* x4 = T4b;
    // up1 @ 8^3 (256 + skip128 -> 256)
    deconv_run(stream, x4, Wu1u, bnq(154), M3, CC3, 3, 256, 256, 384);
    copy_skip<<<cdiv((long)V3 * 128, 256), 256, 0, stream>>>(x3, CC3, V3, 128, 384, 256);
    conv_run(stream, CC3, Wu1rd, bnq(137), M3, nullptr, T3a, 3, 3, 384, 256, 1, 1, 0, 0, 0, 256);
    conv_run(stream, CC3, Wu1ra, bnq(129), M3, nullptr, T3c, 3, 3, 384, 256, 3, 1, 1, 1, 0, 256);
    conv_run(stream, T3c, Wu1rb, bnq(133), M3, T3a, T3d, 3, 3, 256, 256, 3, 1, 1, 1, 0, 256);
    conv_run(stream, T3d, Wu1sa, bnq(144), M3, nullptr, T3a, 3, 3, 256, 256, 3, 1, 1, 1, 0, 256);
    conv_run(stream, T3a, Wu1sb, bnq(148), M3, T3d, T3c, 3, 3, 256, 256, 3, 1, 1, 1, 0, 256);
    h16* y3 = T3c;
    // up2 @ 16^3 (128 + skip64 -> 128)
    deconv_run(stream, y3, Wu2u, bnq(184), M2, CC2, 4, 256, 128, 192);
    copy_skip<<<cdiv((long)V2 * 64, 256), 256, 0, stream>>>(x2, CC2, V2, 64, 192, 128);
    conv_run(stream, CC2, Wu2rd, bnq(167), M2, nullptr, T2a, 4, 4, 192, 128, 1, 1, 0, 0, 0, 128);
    conv_run(stream, CC2, Wu2ra, bnq(159), M2, nullptr, T2c, 4, 4, 192, 128, 3, 1, 1, 1, 0, 128);
    conv_run(stream, T2c, Wu2rb, bnq(163), M2, T2a, T2d, 4, 4, 128, 128, 3, 1, 1, 1, 0, 128);
    conv_run(stream, T2d, Wu2sa, bnq(174), M2, nullptr, T2a, 4, 4, 128, 128, 3, 1, 1, 1, 0, 128);
    conv_run(stream, T2a, Wu2sb, bnq(178), M2, T2d, T2c, 4, 4, 128, 128, 3, 1, 1, 1, 0, 128);
    h16* y2 = T2c;
    // up3 @ 32^3 (96 + skip32 -> 96)
    deconv_run(stream, y2, Wu3u, bnq(214), M1, CC1, 5, 128, 96, 128);
    copy_skip<<<cdiv((long)V1 * 32, 256), 256, 0, stream>>>(x1, CC1, V1, 32, 128, 96);
    conv_run(stream, CC1, Wu3rd, bnq(197), M1, nullptr, T1a, 5, 5, 128, 96, 1, 1, 0, 0, 0, 96);
    conv_run(stream, CC1, Wu3ra, bnq(189), M1, nullptr, T1c, 5, 5, 128, 96, 3, 1, 1, 1, 0, 96);
    conv_run(stream, T1c, Wu3rb, bnq(193), M1, T1a, T1d, 5, 5, 96, 96, 3, 1, 1, 1, 0, 96);
    conv_run(stream, T1d, Wu3sa, bnq(204), M1, nullptr, T1a, 5, 5, 96, 96, 3, 1, 1, 1, 0, 96);
    conv_run(stream, T1a, Wu3sb, bnq(208), M1, T1d, T1c, 5, 5, 96, 96, 3, 1, 1, 1, 0, 96);
    h16* y1 = T1c;
    // up4 @ 64^3 (96 + skip32 -> 96)
    deconv_run(stream, y1, Wu4u, bnq(244), M0, CC0, 6, 96, 96, 128);
    copy_skip<<<cdiv((long)V0 * 32, 256), 256, 0, stream>>>(X0, CC0, V0, 32, 128, 96);
    conv_run(stream, CC0, Wu4rd, bnq(227), M0, nullptr, T0a, 6, 6, 128, 96, 1, 1, 0, 0, 0, 96);
    conv_run(stream, CC0, Wu4ra, bnq(219), M0, nullptr, T0b, 6, 6, 128, 96, 3, 1, 1, 1, 0, 96);
    conv_run(stream, T0b, Wu4rb, bnq(223), M0, T0a, T0c, 6, 6, 96, 96, 3, 1, 1, 1, 0, 96);
    conv_run(stream, T0c, Wu4sa, bnq(234), M0, nullptr, T0a, 6, 6, 96, 96, 3, 1, 1, 1, 0, 96);
    conv_run(stream, T0a, Wu4sb, bnq(238), M0, T0c, T0b, 6, 6, 96, 96, 3, 1, 1, 1, 0, 96);
    // classifier: (y @ W + b) * m0, f32 output (Co padded 32, real 20)
    BN4 cb{nullptr, F(2), nullptr, nullptr};
    conv_run(stream, T0b, Wcls, cb, M0, nullptr, d_out, 6, 6, 96, 32, 1, 1, 0, 0, 1, 20);
}
